// TransformerEncoderLayer_28389733827025
// MI455X (gfx1250) — matrix operand order repaired and checked by reading, not compiled, not run on hardware
//
#include <hip/hip_runtime.h>
#include <hip/hip_bf16.h>

typedef __attribute__((ext_vector_type(16))) _Float16 v16h;
typedef __attribute__((ext_vector_type(8)))  _Float16 v8h;
typedef __attribute__((ext_vector_type(8)))  float    v8f;

#define NB 8
#define SS 512
#define DD 768
#define HH 12
#define HD 64
#define FF 3072
#define BH (NB*HH)
#define BS (NB*SS)

// ---------------------------------------------------------------------------
// WMMA GEMM tile:  C[128x64 tile] = A(MxK,row) * op(B) + epilogue
//   BT=true : B is [N x K] row-major (vector row-copy into LDS) -- weights are
//             pre-transposed so all heavy GEMMs take this path
//   BT=false: B is [K x N] row-major (scalar transpose-on-store; k_o only)
// 256 threads = 8 waves; wave w -> 32x32 sub-tile = 2x2 WMMA 16x16x32 f16.
// ---------------------------------------------------------------------------
template<bool BT>
__device__ __forceinline__ void gemm_tile(
    const _Float16* __restrict__ A, const _Float16* __restrict__ B,
    int K, int lda, int ldb,
    _Float16* __restrict__ outH, float* __restrict__ outF, int ldc,
    const float* __restrict__ bias, const float* __restrict__ resid, int ldr,
    float scale, int act)
{
  const int m0   = blockIdx.x * 128;
  const int n0   = blockIdx.y * 64;
  const int tid  = threadIdx.x;       // 256
  const int wave = tid >> 5;          // 0..7
  const int lane = tid & 31;
  const int wm   = (wave >> 1) * 32;  // 0,32,64,96
  const int wn   = (wave & 1)  * 32;  // 0,32

  __shared__ _Float16 ldsB[64][40];   // [n][k], +8 halfs pad keeps 16B align

  const int rA = lane & 15;           // fragment row / column within 16
  const int kA = (lane >> 4) << 3;    // A frag k-offset: 0 or 8
  const int kB = (lane >> 4) << 3;    // B frag k-offset: 0 or 8

  v8f acc[2][2] = {};

  for (int k0 = 0; k0 < K; k0 += 32) {
    // ---- stage B tile into ldsB[n][k] ----
    if (BT) {
      // B is [N x K]: straight vector row copy (one v8h per thread)
      const int r  = tid >> 2;           // n within tile (0..63)
      const int kk = (tid & 3) * 8;      // 0,8,16,24
      *(v8h*)&ldsB[r][kk] = *(const v8h*)(B + (size_t)(n0 + r) * ldb + k0 + kk);
    } else {
      // B is [K x N]: transpose-on-store (only the small attn*v GEMM)
      const int kr = tid >> 3;           // 0..31
      const int c0 = (tid & 7) * 8;      // 0..56
      const _Float16* src = B + (size_t)(k0 + kr) * ldb + n0 + c0;
      #pragma unroll
      for (int j = 0; j < 8; ++j) ldsB[c0 + j][kr] = src[j];
    }
    __syncthreads();

    if (k0 + 32 < K)  // near-prefetch next A slab
      __builtin_prefetch(A + (size_t)(m0 + wm + rA) * lda + k0 + 32, 0, 3);

    // ---- A fragments (direct from global; per-lane two 16B chunks) ----
    v16h af[2], bf[2];
    #pragma unroll
    for (int i = 0; i < 2; ++i) {
      const _Float16* pa = A + (size_t)(m0 + wm + i * 16 + rA) * lda + k0 + kA;
      v8h lo = *(const v8h*)(pa);
      v8h hi = *(const v8h*)(pa + 16);
      af[i] = __builtin_shufflevector(lo, hi, 0,1,2,3,4,5,6,7,8,9,10,11,12,13,14,15);
    }
    // ---- B fragments from LDS (two 8-half runs per lane) ----
    #pragma unroll
    for (int j = 0; j < 2; ++j) {
      const _Float16* pb = &ldsB[wn + j * 16 + rA][kB];
      v8h lo = *(const v8h*)(pb);
      v8h hi = *(const v8h*)(pb + 16);
      bf[j] = __builtin_shufflevector(lo, hi, 0,1,2,3,4,5,6,7,8,9,10,11,12,13,14,15);
    }
    #pragma unroll
    for (int i = 0; i < 2; ++i)
      #pragma unroll
      for (int j = 0; j < 2; ++j)
        acc[i][j] = __builtin_amdgcn_wmma_f32_16x16x32_f16(
            false, af[i], false, bf[j], (short)0, acc[i][j], false, false);
    __syncthreads();
  }

  // ---- epilogue: scale, +bias, gelu, +residual, store f16 and/or f32 ----
  const int rofs = (lane >> 4) * 8;
  #pragma unroll
  for (int i = 0; i < 2; ++i) {
    #pragma unroll
    for (int j = 0; j < 2; ++j) {
      const int c  = n0 + wn + j * 16 + (lane & 15);
      const float bv = bias ? bias[c] : 0.f;
      #pragma unroll
      for (int r = 0; r < 8; ++r) {
        const int m = m0 + wm + i * 16 + rofs + r;
        float v = acc[i][j][r] * scale + bv;
        if (act) v = 0.5f * v * (1.f + erff(v * 0.70710678118654752f)); // exact GELU
        if (resid) v += resid[(size_t)m * ldr + c];
        if (outF) outF[(size_t)m * ldc + c] = v;
        if (outH) outH[(size_t)m * ldc + c] = (_Float16)v;
      }
    }
  }
}

// ---------------------------------------------------------------------------
// GEMM wrapper kernels (pointer math per blockIdx.z)
// ---------------------------------------------------------------------------
__global__ __launch_bounds__(256) void k_proj(const _Float16* __restrict__ h16,
    const _Float16* __restrict__ WT, const float* __restrict__ bias,
    _Float16* __restrict__ out) {
  const int z = blockIdx.z, b = z / HH, h = z % HH;
  // WT layout: [h][e][d]  (N=64 rows of K=768)
  gemm_tile<true>(h16 + (size_t)b * SS * DD, WT + (size_t)h * HD * DD,
                  DD, DD, DD,
                  out + (size_t)z * SS * HD, nullptr, HD,
                  bias + h * HD, nullptr, 0, 1.f, 0);
}

__global__ __launch_bounds__(256) void k_scores(const _Float16* __restrict__ k16,
    const _Float16* __restrict__ q16, float* __restrict__ scT) {
  const int z = blockIdx.z;
  gemm_tile<true>(k16 + (size_t)z * SS * HD, q16 + (size_t)z * SS * HD,
                  HD, HD, HD,
                  nullptr, scT + (size_t)z * SS * SS, SS,
                  nullptr, nullptr, 0, 0.04419417382415922f /*1/sqrt(512)*/, 0);
}

__global__ __launch_bounds__(256) void k_o(const _Float16* __restrict__ P,
    const _Float16* __restrict__ q16, _Float16* __restrict__ o16) {
  const int z = blockIdx.z, b = z / HH, h = z % HH;
  gemm_tile<false>(P + (size_t)z * SS * SS, q16 + (size_t)z * SS * HD,
                   SS, SS, HD,
                   o16 + (size_t)b * SS * DD + h * HD, nullptr, DD,
                   nullptr, nullptr, 0, 1.f, 0);
}

__global__ __launch_bounds__(256) void k_wo(const _Float16* __restrict__ o16,
    const _Float16* __restrict__ WoT, const float* __restrict__ bo,
    const float* __restrict__ h32, float* __restrict__ x2) {
  const int b = blockIdx.z;
  gemm_tile<true>(o16 + (size_t)b * SS * DD, WoT,
                  DD, DD, DD,
                  nullptr, x2 + (size_t)b * SS * DD, DD,
                  bo, h32 + (size_t)b * SS * DD, DD, 1.f, 0);
}

__global__ __launch_bounds__(256) void k_ff1(const _Float16* __restrict__ h2,
    const _Float16* __restrict__ W1T, const float* __restrict__ b1,
    _Float16* __restrict__ ff16) {
  const int b = blockIdx.z;
  gemm_tile<true>(h2 + (size_t)b * SS * DD, W1T,
                  DD, DD, DD,
                  ff16 + (size_t)b * SS * FF, nullptr, FF,
                  b1, nullptr, 0, 1.f, 1 /*gelu*/);
}

__global__ __launch_bounds__(256) void k_ff2(const _Float16* __restrict__ ff16,
    const _Float16* __restrict__ W2T, const float* __restrict__ b2,
    const float* __restrict__ x2, float* __restrict__ out) {
  const int b = blockIdx.z;
  gemm_tile<true>(ff16 + (size_t)b * SS * FF, W2T,
                  FF, FF, FF,
                  nullptr, out + (size_t)b * SS * DD, DD,
                  b2, x2 + (size_t)b * SS * DD, DD, 1.f, 0);
}

// ---------------------------------------------------------------------------
// LayerNorm over 768-wide rows; writes f32 (optional) + f16
// ---------------------------------------------------------------------------
__global__ __launch_bounds__(256) void k_ln(const float* __restrict__ x,
    const float* __restrict__ g, const float* __restrict__ be,
    float* __restrict__ o32, _Float16* __restrict__ o16) {
  const int row = blockIdx.x, tid = threadIdx.x;
  const float* xr = x + (size_t)row * DD;
  float v0 = xr[tid], v1 = xr[tid + 256], v2 = xr[tid + 512];
  __shared__ float red[256];
  red[tid] = v0 + v1 + v2; __syncthreads();
  for (int o = 128; o > 0; o >>= 1) { if (tid < o) red[tid] += red[tid + o]; __syncthreads(); }
  const float mean = red[0] * (1.f / 768.f);
  __syncthreads();
  const float d0 = v0 - mean, d1 = v1 - mean, d2 = v2 - mean;
  red[tid] = d0 * d0 + d1 * d1 + d2 * d2; __syncthreads();
  for (int o = 128; o > 0; o >>= 1) { if (tid < o) red[tid] += red[tid + o]; __syncthreads(); }
  const float rstd = rsqrtf(red[0] * (1.f / 768.f) + 1e-5f);
  const float h0 = d0 * rstd * g[tid]       + be[tid];
  const float h1 = d1 * rstd * g[tid + 256] + be[tid + 256];
  const float h2 = d2 * rstd * g[tid + 512] + be[tid + 512];
  const size_t base = (size_t)row * DD;
  if (o32) { o32[base + tid] = h0; o32[base + tid + 256] = h1; o32[base + tid + 512] = h2; }
  o16[base + tid] = (_Float16)h0; o16[base + tid + 256] = (_Float16)h1; o16[base + tid + 512] = (_Float16)h2;
}

// transpose-convert: in[bz][r][c] f32 -> out[bz][c][r] f16 (R,C mult of 32)
__global__ __launch_bounds__(256) void k_cvtT(const float* __restrict__ in,
    _Float16* __restrict__ outp, int R, int C) {
  __shared__ float tile[32][33];
  const int bz = blockIdx.z;
  const int r0 = blockIdx.x * 32, c0 = blockIdx.y * 32;
  const int tx = threadIdx.x, ty = threadIdx.y;  // (32, 8)
  const float* src = in + (size_t)bz * R * C;
  _Float16* dst = outp + (size_t)bz * R * C;
  #pragma unroll
  for (int i = 0; i < 4; ++i)
    tile[ty + 8 * i][tx] = src[(size_t)(r0 + ty + 8 * i) * C + c0 + tx];
  __syncthreads();
  #pragma unroll
  for (int i = 0; i < 4; ++i)
    dst[(size_t)(c0 + ty + 8 * i) * R + r0 + tx] = (_Float16)tile[tx][ty + 8 * i];
}

// softmax pass 1: per row t of scoresT -> {max, 1/sum}
__global__ __launch_bounds__(256) void k_soft1(const float* __restrict__ sc,
    float2* __restrict__ ms) {
  const int row = blockIdx.x, tid = threadIdx.x;
  const float* r = sc + (size_t)row * SS;
  const float a = r[tid], b = r[tid + 256];
  __shared__ float red[256];
  red[tid] = fmaxf(a, b); __syncthreads();
  for (int o = 128; o > 0; o >>= 1) { if (tid < o) red[tid] = fmaxf(red[tid], red[tid + o]); __syncthreads(); }
  const float mx = red[0]; __syncthreads();
  red[tid] = __expf(a - mx) + __expf(b - mx); __syncthreads();
  for (int o = 128; o > 0; o >>= 1) { if (tid < o) red[tid] += red[tid + o]; __syncthreads(); }
  if (tid == 0) ms[row] = make_float2(mx, 1.f / red[0]);
}

// softmax pass 2: transpose-exp  P[s][t] = exp(scT[t][s]-mx[t]) * inv[t]  (f16)
__global__ __launch_bounds__(256) void k_soft2(const float* __restrict__ sc,
    const float2* __restrict__ ms, _Float16* __restrict__ P) {
  __shared__ float tile[32][33];
  const int bh = blockIdx.z, t0 = blockIdx.x * 32, s0 = blockIdx.y * 32;
  const int tx = threadIdx.x, ty = threadIdx.y;  // (32, 8)
  const float* src = sc + (size_t)bh * SS * SS;
  #pragma unroll
  for (int i = 0; i < 4; ++i)
    tile[ty + 8 * i][tx] = src[(size_t)(t0 + ty + 8 * i) * SS + s0 + tx];
  __syncthreads();
  const float2 m = ms[bh * SS + t0 + tx];
  _Float16* dst = P + (size_t)bh * SS * SS;
  #pragma unroll
  for (int i = 0; i < 4; ++i) {
    const float v = tile[tx][ty + 8 * i];
    dst[(size_t)(s0 + ty + 8 * i) * SS + t0 + tx] = (_Float16)(__expf(v - m.x) * m.y);
  }
}

// ---------------------------------------------------------------------------
extern "C" void kernel_launch(void* const* d_in, const int* in_sizes, int n_in,
                              void* d_out, int out_size, void* d_ws, size_t ws_size,
                              hipStream_t stream) {
  const float* x    = (const float*)d_in[0];
  const float* Wq   = (const float*)d_in[1];
  const float* bq   = (const float*)d_in[2];
  const float* Wk   = (const float*)d_in[3];
  const float* bk   = (const float*)d_in[4];
  // d_in[5]=Wv, d_in[6]=bv intentionally unused (reference bug: v = q)
  const float* Wo   = (const float*)d_in[7];
  const float* bo   = (const float*)d_in[8];
  const float* g1   = (const float*)d_in[9];
  const float* be1  = (const float*)d_in[10];
  const float* g2   = (const float*)d_in[11];
  const float* be2  = (const float*)d_in[12];
  const float* W1   = (const float*)d_in[13];
  const float* b1   = (const float*)d_in[14];
  const float* W2   = (const float*)d_in[15];
  const float* b2   = (const float*)d_in[16];
  float* out = (float*)d_out;

  char* ws = (char*)d_ws;
  size_t off = 0;
  auto alloc = [&](size_t bytes) { char* p = ws + off; off += (bytes + 255) & ~(size_t)255; return p; };

  float*    h32  = (float*)   alloc((size_t)BS * DD * 4);      // post-LN1 (residual)
  _Float16* h16  = (_Float16*)alloc((size_t)BS * DD * 2);      // reused as o16
  _Float16* WqT  = (_Float16*)alloc((size_t)HH * DD * HD * 2); // [h][e][d]
  _Float16* WkT  = (_Float16*)alloc((size_t)HH * DD * HD * 2);
  _Float16* WoT  = (_Float16*)alloc((size_t)DD * DD * 2);      // [j][i]
  _Float16* W1T  = (_Float16*)alloc((size_t)DD * FF * 2);      // [F][D]
  _Float16* W2T  = (_Float16*)alloc((size_t)FF * DD * 2);      // [D][F]
  _Float16* q16  = (_Float16*)alloc((size_t)BH * SS * HD * 2); // reused as h2
  _Float16* k16  = (_Float16*)alloc((size_t)BH * SS * HD * 2);
  float*    scT  = (float*)   alloc((size_t)BH * SS * SS * 4); // reused as ff16
  _Float16* P    = (_Float16*)alloc((size_t)BH * SS * SS * 2);
  float2*   ms   = (float2*)  alloc((size_t)BH * SS * 8);
  float*    x2   = (float*)   alloc((size_t)BS * DD * 4);
  // buffer reuse (lifetimes disjoint):
  _Float16* o16  = h16;                 // h16 dead after q/k projections
  _Float16* h2   = q16;                 // q16 dead after k_o
  _Float16* ff16 = (_Float16*)scT;      // scoresT dead after soft2

  // 1) LN1
  k_ln<<<BS, 256, 0, stream>>>(x, g1, be1, h32, h16);
  // 2) weight transpose-converts f32 -> f16 (so all weight GEMMs are BT mode)
  dim3 tb(32, 8);
  k_cvtT<<<dim3(DD/32, HD/32, HH), tb, 0, stream>>>(Wq, WqT, DD, HD);
  k_cvtT<<<dim3(DD/32, HD/32, HH), tb, 0, stream>>>(Wk, WkT, DD, HD);
  k_cvtT<<<dim3(DD/32, DD/32, 1),  tb, 0, stream>>>(Wo, WoT, DD, DD);
  k_cvtT<<<dim3(DD/32, FF/32, 1),  tb, 0, stream>>>(W1, W1T, DD, FF);
  k_cvtT<<<dim3(FF/32, DD/32, 1),  tb, 0, stream>>>(W2, W2T, FF, DD);
  // 3) q/k projections  [per (b,h): 512x768 * (64x768)^T]
  k_proj<<<dim3(4, 1, BH), 256, 0, stream>>>(h16, WqT, bq, q16);
  k_proj<<<dim3(4, 1, BH), 256, 0, stream>>>(h16, WkT, bk, k16);
  // 4) scoresT[t][s] = k.qT / sqrt(512)   [per (b,h): 512x64 * (512x64)^T]
  k_scores<<<dim3(4, 8, BH), 256, 0, stream>>>(k16, q16, scT);
  // 5) softmax over query axis == row softmax of scoresT; emit attn[s][t] f16
  k_soft1<<<BH * SS, 256, 0, stream>>>(scT, ms);
  k_soft2<<<dim3(16, 16, BH), tb, 0, stream>>>(scT, ms, P);
  // 6) o = attn * q (v=q bug), heads concatenated via column offset
  k_o<<<dim3(4, 1, BH), 256, 0, stream>>>(P, q16, o16);
  // 7) x2 = h + o*Wo + bo
  k_wo<<<dim3(4, HH, NB), 256, 0, stream>>>(o16, WoT, bo, h32, x2);
  // 8) LN2
  k_ln<<<BS, 256, 0, stream>>>(x2, g2, be2, nullptr, h2);
  // 9) ff = gelu(h2*W1 + b1)
  k_ff1<<<dim3(4, FF / 64, NB), 256, 0, stream>>>(h2, W1T, b1, ff16);
  // 10) out = x2 + ff*W2 + b2
  k_ff2<<<dim3(4, HH, NB), 256, 0, stream>>>(ff16, W2T, b2, x2, out);

  (void)in_sizes; (void)n_in; (void)out_size; (void)ws_size;
}
